// ViT_3332894621993
// MI455X (gfx1250) — matrix operand order repaired and checked by reading, not compiled, not run on hardware
//
#include <hip/hip_runtime.h>
#include <hip/hip_bf16.h>
#include <math.h>

// ---------------------------------------------------------------------------
// Types for CDNA5 WMMA (wave32). 16x16x32 bf16 -> f32 accumulate.
// ---------------------------------------------------------------------------
typedef __attribute__((ext_vector_type(16))) __bf16        v16bf;
typedef __attribute__((ext_vector_type(2)))  __bf16        v2bf;
typedef __attribute__((ext_vector_type(8)))  float         v8f;
typedef __attribute__((ext_vector_type(8)))  unsigned int  v8u;

#define EPI_BIAS 1
#define EPI_GELU 2
#define EPI_RES  4

template <bool V> struct BoolC { static constexpr bool value = V; };

__device__ __forceinline__ unsigned int pack_bf2(float a, float b) {
#if __has_builtin(__builtin_amdgcn_cvt_pk_bf16_f32)
  v2bf r = __builtin_amdgcn_cvt_pk_bf16_f32(a, b);
  return __builtin_bit_cast(unsigned int, r);
#else
  // RNE: bias by 0x7fff + lsb-of-result, then take the two high halves
  unsigned int ua = __float_as_uint(a), ub = __float_as_uint(b);
  unsigned int ta = ua + 0x7fffu + ((ua >> 16) & 1u);
  unsigned int tb = ub + 0x7fffu + ((ub >> 16) & 1u);
  return __builtin_amdgcn_perm(tb, ta, 0x07060302u);  // [tb.hi16 : ta.hi16]
#endif
}

__device__ __forceinline__ float gelu_exact(float v) {
  return 0.5f * v * (1.0f + erff(v * 0.70710678118654752f));
}

// ---------------------------------------------------------------------------
// Generic batched GEMM:  D[m,n] = sum_k A[m,k] * B[k,n]  (+ epilogue)
//   A addressed as A[m*lda + k]            (+ bz*sA)
//   B addressed as B[k*ldbK + n*ldbN]      (+ bz*sB)   (stride-swap => B^T)
//   D addressed as D[m*ldd + n]            (+ bz*sD)
// 128 threads (4 waves), 128x64 block tile, K-step 32. Wave w owns a 32x64
// slab: 2 A-frags + 4 shared B-frags -> 8 WMMAs per step (3 DS-instr/WMMA).
// Double-buffered LDS, one barrier per K-step. Fast path: b128 coalesced
// global loads; K-tail handled separately so only true MN-edges take the
// fully guarded path.
// ---------------------------------------------------------------------------
template <int EPI>
__global__ __launch_bounds__(128)
void gemm_wmma(const float* __restrict__ A, const float* __restrict__ Bm,
               const float* __restrict__ bias, const float* __restrict__ res,
               float* __restrict__ D,
               int M, int N, int K,
               int lda, int ldbK, int ldbN, int ldd,
               long long sA, long long sB, long long sD)
{
  __shared__ unsigned int As[2][128][17];  // [buf][row m][k-pair] (padded)
  __shared__ unsigned int Bs[2][64][17];   // [buf][col n][k-pair] (B transposed)

  const long long bz = blockIdx.z;
  A  += bz * sA;
  Bm += bz * sB;
  D  += bz * sD;
  const float* R = (EPI & EPI_RES) ? (res + bz * sD) : nullptr;

  const int tid  = threadIdx.x;
  const int wv   = tid >> 5;
  const int lane = tid & 31;
  const int hl   = lane >> 4;      // lane half (ISA fragment layout)
  const int l15  = lane & 15;
  const int tileM = blockIdx.y * 128;
  const int tileN = blockIdx.x * 64;
  const bool fast = (tileM + 128 <= M) && (tileN + 64 <= N);
  const int nk = (K + 31) >> 5;

  v8f acc[8];
#pragma unroll
  for (int i = 0; i < 8; ++i)
    acc[i] = v8f{0.f, 0.f, 0.f, 0.f, 0.f, 0.f, 0.f, 0.f};

  auto run = [&](auto gc) {
    constexpr bool G = decltype(gc)::value;
    // register staging: 128x32 A tile (32 f/thread) + 32x64 B tile (16 f/thread)
    float aR[32];
    float bR[16];

    auto load_tile = [&](int k0, bool kt) {
      if constexpr (!G) {
        // ---- A: float4 along K (contiguous); 8 threads = 128B/row ----
#pragma unroll
        for (int i = 0; i < 8; ++i) {
          int idx = tid + i * 128;           // 0..1023
          int row = idx >> 3;                // 0..127
          int q   = idx & 7;                 // float4 slot (k = 4q)
          const float* ap = A + (long long)(tileM + row) * lda + k0 + q * 4;
          float4 v;
          if (!kt) {
            v = *(const float4*)ap;
          } else {
            int gk = k0 + q * 4;
            v.x = (gk     < K) ? ap[0] : 0.f;
            v.y = (gk + 1 < K) ? ap[1] : 0.f;
            v.z = (gk + 2 < K) ? ap[2] : 0.f;
            v.w = (gk + 3 < K) ? ap[3] : 0.f;
          }
          aR[i * 4 + 0] = v.x; aR[i * 4 + 1] = v.y;
          aR[i * 4 + 2] = v.z; aR[i * 4 + 3] = v.w;
        }
        if (ldbN == 1) {
          // ---- B row-major: float4 along N (contiguous), rows k,k+1 ----
#pragma unroll
          for (int i = 0; i < 2; ++i) {
            int idx = tid + i * 128;         // 0..255
            int kk  = idx >> 4;              // k-pair 0..15
            int n4  = (idx & 15) * 4;        // 0,4,..,60
            int gk  = k0 + kk * 2;
            const float* bp = Bm + (long long)gk * ldbK + (tileN + n4);
            float4 r0, r1;
            if (!kt) {
              r0 = *(const float4*)bp;
              r1 = *(const float4*)(bp + ldbK);
            } else {
              r0 = (gk     < K) ? *(const float4*)bp
                                : float4{0.f, 0.f, 0.f, 0.f};
              r1 = (gk + 1 < K) ? *(const float4*)(bp + ldbK)
                                : float4{0.f, 0.f, 0.f, 0.f};
            }
            bR[i * 8 + 0] = r0.x; bR[i * 8 + 1] = r1.x;
            bR[i * 8 + 2] = r0.y; bR[i * 8 + 3] = r1.y;
            bR[i * 8 + 4] = r0.z; bR[i * 8 + 5] = r1.z;
            bR[i * 8 + 6] = r0.w; bR[i * 8 + 7] = r1.w;
          }
        } else {
          // ---- B K-contiguous (ldbK==1, Q.K^T): float4 along K ----
#pragma unroll
          for (int i = 0; i < 4; ++i) {
            int idx = tid + i * 128;         // 0..511
            int n   = idx >> 3;              // 0..63
            int q   = idx & 7;
            const float* bp = Bm + (long long)(tileN + n) * ldbN + k0 + q * 4;
            float4 v;
            if (!kt) {
              v = *(const float4*)bp;
            } else {
              int gk = k0 + q * 4;
              v.x = (gk     < K) ? bp[0] : 0.f;
              v.y = (gk + 1 < K) ? bp[1] : 0.f;
              v.z = (gk + 2 < K) ? bp[2] : 0.f;
              v.w = (gk + 3 < K) ? bp[3] : 0.f;
            }
            bR[i * 4 + 0] = v.x; bR[i * 4 + 1] = v.y;
            bR[i * 4 + 2] = v.z; bR[i * 4 + 3] = v.w;
          }
        }
      } else {
        // ---- guarded edge path (rare): scalar loads, zero-fill OOB ----
#pragma unroll
        for (int i = 0; i < 16; ++i) {       // A: 128 rows x 16 k-pairs
          int idx = tid + i * 128;           // 0..2047
          int row = idx >> 4;                // 0..127
          int kp  = idx & 15;
          int gk  = k0 + kp * 2;
          int gr  = tileM + row;
          float a0 = 0.f, a1 = 0.f;
          if (gr < M) {
            if (gk     < K) a0 = A[(long long)gr * lda + gk];
            if (gk + 1 < K) a1 = A[(long long)gr * lda + gk + 1];
          }
          aR[i * 2] = a0; aR[i * 2 + 1] = a1;
        }
#pragma unroll
        for (int i = 0; i < 8; ++i) {        // B: 64 cols x 16 k-pairs
          int idx = tid + i * 128;           // 0..1023
          int col = idx >> 4;
          int kp  = idx & 15;
          int gk  = k0 + kp * 2;
          int gn  = tileN + col;
          float b0 = 0.f, b1 = 0.f;
          if (gn < N) {
            if (gk     < K) b0 = Bm[(long long)gk * ldbK + (long long)gn * ldbN];
            if (gk + 1 < K) b1 = Bm[(long long)(gk + 1) * ldbK + (long long)gn * ldbN];
          }
          bR[i * 2] = b0; bR[i * 2 + 1] = b1;
        }
      }
      // prefetch the tile after next (emits global_prefetch_b8)
      if (k0 + 64 < K) {
        if (tileM + tid < M)
          __builtin_prefetch(A + (long long)(tileM + tid) * lda + (k0 + 64));
      }
    };

    auto store_tile = [&](int buf) {
      if constexpr (!G) {
#pragma unroll
        for (int i = 0; i < 8; ++i) {
          int idx = tid + i * 128;
          int row = idx >> 3;
          int q   = idx & 7;
          As[buf][row][2 * q]     = pack_bf2(aR[i * 4 + 0], aR[i * 4 + 1]);
          As[buf][row][2 * q + 1] = pack_bf2(aR[i * 4 + 2], aR[i * 4 + 3]);
        }
        if (ldbN == 1) {
#pragma unroll
          for (int i = 0; i < 2; ++i) {
            int idx = tid + i * 128;
            int kk  = idx >> 4;
            int n4  = (idx & 15) * 4;
            Bs[buf][n4 + 0][kk] = pack_bf2(bR[i * 8 + 0], bR[i * 8 + 1]);
            Bs[buf][n4 + 1][kk] = pack_bf2(bR[i * 8 + 2], bR[i * 8 + 3]);
            Bs[buf][n4 + 2][kk] = pack_bf2(bR[i * 8 + 4], bR[i * 8 + 5]);
            Bs[buf][n4 + 3][kk] = pack_bf2(bR[i * 8 + 6], bR[i * 8 + 7]);
          }
        } else {
#pragma unroll
          for (int i = 0; i < 4; ++i) {
            int idx = tid + i * 128;
            int n   = idx >> 3;
            int q   = idx & 7;
            Bs[buf][n][2 * q]     = pack_bf2(bR[i * 4 + 0], bR[i * 4 + 1]);
            Bs[buf][n][2 * q + 1] = pack_bf2(bR[i * 4 + 2], bR[i * 4 + 3]);
          }
        }
      } else {
#pragma unroll
        for (int i = 0; i < 16; ++i) {
          int idx = tid + i * 128;
          int row = idx >> 4;
          int kp  = idx & 15;
          As[buf][row][kp] = pack_bf2(aR[i * 2], aR[i * 2 + 1]);
        }
#pragma unroll
        for (int i = 0; i < 8; ++i) {
          int idx = tid + i * 128;
          int col = idx >> 4;
          int kp  = idx & 15;
          Bs[buf][col][kp] = pack_bf2(bR[i * 2], bR[i * 2 + 1]);
        }
      }
    };

    load_tile(0, 32 > K);
    store_tile(0);
    __syncthreads();

    for (int it = 0; it < nk; ++it) {
      if (it + 1 < nk) {
        const int k0n = (it + 1) << 5;
        load_tile(k0n, k0n + 32 > K);
      }

      const int buf = it & 1;
      // ---- gather fragments (ISA 7.12.2 layouts), then WMMA chain ----
      v8u au0, au1;
      const int mrow = wv * 32 + l15;
#pragma unroll
      for (int j = 0; j < 8; ++j) {
        const int ki = ((j >> 2) << 3) + (hl << 2) + (j & 3);
        au0[j] = As[buf][mrow][ki];
        au1[j] = As[buf][mrow + 16][ki];
      }

      v16bf av0 = __builtin_bit_cast(v16bf, au0);
      v16bf av1 = __builtin_bit_cast(v16bf, au1);
#pragma unroll
      for (int nn = 0; nn < 4; ++nn) {
        v8u bu;
        const int ncol = nn * 16 + l15;
#pragma unroll
        for (int j = 0; j < 8; ++j)
          bu[j] = Bs[buf][ncol][((j >> 2) << 3) + (hl << 2) + (j & 3)];
        v16bf bv = __builtin_bit_cast(v16bf, bu);
        acc[nn]     = __builtin_amdgcn_wmma_f32_16x16x32_bf16(
            false, av0, false, bv, (short)0, acc[nn], false, false);
        acc[4 + nn] = __builtin_amdgcn_wmma_f32_16x16x32_bf16(
            false, av1, false, bv, (short)0, acc[4 + nn], false, false);
      }

      if (it + 1 < nk) {
        store_tile((it + 1) & 1);
        __syncthreads();
      }
    }
  };

  if (fast) run(BoolC<false>{});
  else      run(BoolC<true>{});

  // ---- epilogue: C/D layout: VGPR r -> (M = r + 8*half, N = l15) ----
#pragma unroll
  for (int mh = 0; mh < 2; ++mh) {
#pragma unroll
    for (int nn = 0; nn < 4; ++nn) {
      int gc = tileN + nn * 16 + l15;
      if (gc >= N) continue;
#pragma unroll
      for (int r = 0; r < 8; ++r) {
        int gr = tileM + wv * 32 + mh * 16 + hl * 8 + r;
        if (gr >= M) continue;
        float v = acc[mh * 4 + nn][r];
        if (EPI & EPI_BIAS) v += bias[gc];
        if (EPI & EPI_GELU) v = gelu_exact(v);
        if (EPI & EPI_RES)  v += R[(long long)gr * ldd + gc];
        D[(long long)gr * ldd + gc] = v;
      }
    }
  }
}

// ---------------------------------------------------------------------------
// LayerNorm over last dim (768). One block per token row, 256 threads.
// ---------------------------------------------------------------------------
__global__ __launch_bounds__(256)
void ln_kernel(const float* __restrict__ x, const float* __restrict__ g,
               const float* __restrict__ b, float* __restrict__ y)
{
  __shared__ float red[256];
  const long long row = blockIdx.x;
  const float* xr = x + row * 768;
  float* yr = y + row * 768;
  const int tid = threadIdx.x;

  float v0 = xr[tid], v1 = xr[tid + 256], v2 = xr[tid + 512];
  red[tid] = v0 + v1 + v2;
  __syncthreads();
  for (int off = 128; off > 0; off >>= 1) {
    if (tid < off) red[tid] += red[tid + off];
    __syncthreads();
  }
  float mean = red[0] * (1.f / 768.f);
  __syncthreads();

  float d0 = v0 - mean, d1 = v1 - mean, d2 = v2 - mean;
  red[tid] = d0 * d0 + d1 * d1 + d2 * d2;
  __syncthreads();
  for (int off = 128; off > 0; off >>= 1) {
    if (tid < off) red[tid] += red[tid + off];
    __syncthreads();
  }
  float inv = rsqrtf(red[0] * (1.f / 768.f) + 1e-5f);

  yr[tid]       = d0 * inv * g[tid]       + b[tid];
  yr[tid + 256] = d1 * inv * g[tid + 256] + b[tid + 256];
  yr[tid + 512] = d2 * inv * g[tid + 512] + b[tid + 512];
}

// ---------------------------------------------------------------------------
// Softmax over rows of length 592 (577 valid, tail zeroed).
// ---------------------------------------------------------------------------
__global__ __launch_bounds__(256)
void softmax_kernel(float* __restrict__ p)
{
  __shared__ float red[256];
  const long long row = blockIdx.x;
  float* pr = p + row * 592;
  const int tid = threadIdx.x;
  const float NEG = -3.0e38f;

  float a0 = (tid       < 577) ? pr[tid]       : NEG;
  float a1 = (tid + 256 < 577) ? pr[tid + 256] : NEG;
  float a2 = (tid + 512 < 577) ? pr[tid + 512] : NEG;

  float mx = fmaxf(a0, fmaxf(a1, a2));
  red[tid] = mx;
  __syncthreads();
  for (int off = 128; off > 0; off >>= 1) {
    if (tid < off) red[tid] = fmaxf(red[tid], red[tid + off]);
    __syncthreads();
  }
  mx = red[0];
  __syncthreads();

  float e0 = (tid       < 577) ? __expf(a0 - mx) : 0.f;
  float e1 = (tid + 256 < 577) ? __expf(a1 - mx) : 0.f;
  float e2 = (tid + 512 < 577) ? __expf(a2 - mx) : 0.f;
  red[tid] = e0 + e1 + e2;
  __syncthreads();
  for (int off = 128; off > 0; off >>= 1) {
    if (tid < off) red[tid] += red[tid + off];
    __syncthreads();
  }
  float inv = 1.f / red[0];

  pr[tid]       = e0 * inv;
  pr[tid + 256] = e1 * inv;
  if (tid + 512 < 592) pr[tid + 512] = e2 * inv;   // pad cols -> 0
}

// ---------------------------------------------------------------------------
// img (8,3,384,384) -> patches (8*576, 768) with (p1,p2,c) fastest order.
// ---------------------------------------------------------------------------
__global__ __launch_bounds__(256)
void patchify_kernel(const float* __restrict__ img, float* __restrict__ patches)
{
  long long idx = (long long)blockIdx.x * 256 + threadIdx.x;
  const long long total = 8LL * 576 * 768;
  if (idx >= total) return;
  int d = idx % 768;
  int p = (idx / 768) % 576;
  int b = idx / (768LL * 576);
  int c  = d % 3;
  int p2 = (d / 3) % 16;
  int p1 = d / 48;
  int wx = p % 24;
  int hy = p / 24;
  int row = hy * 16 + p1;
  int col = wx * 16 + p2;
  patches[idx] = img[(((long long)b * 3 + c) * 384 + row) * 384 + col];
}

// x[b,t,:] = pos[t,:] + (t==0 ? cls : patch_embed[b, t-1, :])
__global__ __launch_bounds__(256)
void assemble_kernel(const float* __restrict__ pe, const float* __restrict__ cls,
                     const float* __restrict__ pos, float* __restrict__ x)
{
  long long idx = (long long)blockIdx.x * 256 + threadIdx.x;
  const long long total = 8LL * 577 * 768;
  if (idx >= total) return;
  int d = idx % 768;
  int t = (idx / 768) % 577;
  int b = idx / (768LL * 577);
  float v = pos[(long long)t * 768 + d];
  v += (t == 0) ? cls[d] : pe[((long long)b * 576 + (t - 1)) * 768 + d];
  x[idx] = v;
}

// out = x[:, 1:, :]
__global__ __launch_bounds__(256)
void strip_cls_kernel(const float* __restrict__ x, float* __restrict__ out)
{
  long long idx = (long long)blockIdx.x * 256 + threadIdx.x;
  const long long total = 8LL * 576 * 768;
  if (idx >= total) return;
  int d = idx % 768;
  int p = (idx / 768) % 576;
  int b = idx / (768LL * 576);
  out[idx] = x[((long long)b * 577 + 1 + p) * 768 + d];
}

// ---------------------------------------------------------------------------
// Host-side orchestration
// ---------------------------------------------------------------------------
extern "C" void kernel_launch(void* const* d_in, const int* in_sizes, int n_in,
                              void* d_out, int out_size, void* d_ws, size_t ws_size,
                              hipStream_t stream)
{
  (void)in_sizes; (void)n_in; (void)out_size; (void)ws_size;

  const float* img   = (const float*)d_in[0];
  const float* pos   = (const float*)d_in[1];
  const float* cls   = (const float*)d_in[2];
  const float* pw    = (const float*)d_in[3];
  const float* pbias = (const float*)d_in[4];
  const float* ln1g  = (const float*)d_in[5];
  const float* ln1b  = (const float*)d_in[6];
  const float* qkvw  = (const float*)d_in[7];
  const float* outw  = (const float*)d_in[8];
  const float* outb  = (const float*)d_in[9];
  const float* ln2g  = (const float*)d_in[10];
  const float* ln2b  = (const float*)d_in[11];
  const float* ff1w  = (const float*)d_in[12];
  const float* ff1b  = (const float*)d_in[13];
  const float* ff2w  = (const float*)d_in[14];
  const float* ff2b  = (const float*)d_in[15];
  float* out = (float*)d_out;

  const int NB = 8, T = 577, DM = 768, H3 = 2304, ML = 3072, NH = 12, DEPTH = 8;
  const int ROWS = NB * T;                 // 4616
  const long long XSZ  = (long long)NB * T * DM;          // x / y / o buffers
  const long long PSZ  = (long long)NH * T * 592;         // per-image padded probs
  float* ws   = (float*)d_ws;
  float* xb   = ws;                        // residual stream (8,577,768)
  float* yb   = ws + XSZ;                  // LN output / patches
  float* ob   = ws + 2 * XSZ;              // attention output
  float* pb   = ws + 3 * XSZ;              // probs (12,577,592)
  float* bigb = ws + 3 * XSZ + PSZ;        // qkv (4616x2304) / ff1 (4616x3072) / patch-embed

  auto cdiv = [](int a, int b) { return (a + b - 1) / b; };

  // ---- patch embedding ----
  {
    long long n = 8LL * 576 * 768;
    patchify_kernel<<<dim3((unsigned)((n + 255) / 256)), 256, 0, stream>>>(img, yb);
    gemm_wmma<EPI_BIAS><<<dim3(cdiv(DM, 64), cdiv(4608, 128), 1), 128, 0, stream>>>(
        yb, pw, pbias, nullptr, bigb,
        4608, DM, DM, DM, DM, 1, DM, 0, 0, 0);
    long long m = 8LL * 577 * 768;
    assemble_kernel<<<dim3((unsigned)((m + 255) / 256)), 256, 0, stream>>>(bigb, cls, pos, xb);
  }

  // ---- transformer layers ----
  for (int l = 0; l < DEPTH; ++l) {
    const float* wqkv = qkvw + (long long)l * DM * H3;
    const float* wo   = outw + (long long)l * DM * DM;
    const float* bo   = outb + (long long)l * DM;
    const float* w1   = ff1w + (long long)l * DM * ML;
    const float* b1   = ff1b + (long long)l * ML;
    const float* w2   = ff2w + (long long)l * ML * DM;
    const float* b2   = ff2b + (long long)l * DM;

    // LN1
    ln_kernel<<<dim3(ROWS), 256, 0, stream>>>(xb, ln1g + (long long)l * DM,
                                              ln1b + (long long)l * DM, yb);
    // qkv = y @ Wqkv  (no bias in reference)
    gemm_wmma<0><<<dim3(cdiv(H3, 64), cdiv(ROWS, 128), 1), 128, 0, stream>>>(
        yb, wqkv, nullptr, nullptr, bigb,
        ROWS, H3, DM, DM, H3, 1, H3, 0, 0, 0);

    // attention, per image, 12 heads batched via grid.z
    for (int b = 0; b < NB; ++b) {
      const float* qkvb = bigb + (long long)b * T * H3;
      // dots[h] = Q_h (577x64) @ K_h^T  -> probs (577x577, ld 592)
      gemm_wmma<0><<<dim3(cdiv(T, 64), cdiv(T, 128), NH), 128, 0, stream>>>(
          qkvb, qkvb + DM, nullptr, nullptr, pb,
          T, T, 64,
          /*lda*/ H3, /*ldbK*/ 1, /*ldbN*/ H3, /*ldd*/ 592,
          /*sA*/ 64, /*sB*/ 64, /*sD*/ (long long)T * 592);
      softmax_kernel<<<dim3(NH * T), 256, 0, stream>>>(pb);
      // o[h] = P (577x577) @ V_h (577x64)
      gemm_wmma<0><<<dim3(1, cdiv(T, 128), NH), 128, 0, stream>>>(
          pb, qkvb + 2 * DM, nullptr, nullptr, ob + (long long)b * T * DM,
          T, 64, T,
          /*lda*/ 592, /*ldbK*/ H3, /*ldbN*/ 1, /*ldd*/ DM,
          /*sA*/ (long long)T * 592, /*sB*/ 64, /*sD*/ 64);
    }

    // x = x + o @ Wo + bo
    gemm_wmma<EPI_BIAS | EPI_RES><<<dim3(cdiv(DM, 64), cdiv(ROWS, 128), 1), 128, 0, stream>>>(
        ob, wo, bo, xb, xb,
        ROWS, DM, DM, DM, DM, 1, DM, 0, 0, 0);

    // LN2
    ln_kernel<<<dim3(ROWS), 256, 0, stream>>>(xb, ln2g + (long long)l * DM,
                                              ln2b + (long long)l * DM, yb);
    // h = gelu(y @ W1 + b1)
    gemm_wmma<EPI_BIAS | EPI_GELU><<<dim3(cdiv(ML, 64), cdiv(ROWS, 128), 1), 128, 0, stream>>>(
        yb, w1, b1, nullptr, bigb,
        ROWS, ML, DM, DM, ML, 1, ML, 0, 0, 0);
    // x = x + h @ W2 + b2
    gemm_wmma<EPI_BIAS | EPI_RES><<<dim3(cdiv(DM, 64), cdiv(ROWS, 128), 1), 128, 0, stream>>>(
        bigb, w2, b2, xb, xb,
        ROWS, DM, ML, ML, DM, 1, DM, 0, 0, 0);
  }

  // ---- drop CLS token ----
  {
    long long n = 8LL * 576 * 768;
    strip_cls_kernel<<<dim3((unsigned)((n + 255) / 256)), 256, 0, stream>>>(xb, out);
  }
}
